// Restormer_35012573397389
// MI455X (gfx1250) — matrix operand order repaired and checked by reading, not compiled, not run on hardware
//
#include <hip/hip_runtime.h>

// ---------------------------------------------------------------------------
// Types for CDNA5 WMMA (wave32, 16x16x32 bf16 -> f32)
// ---------------------------------------------------------------------------
typedef __attribute__((ext_vector_type(16))) __bf16 v16bf;
typedef __attribute__((ext_vector_type(8)))  float  v8f;
typedef float f4u __attribute__((ext_vector_type(4), aligned(4)));  // unaligned-ok float4

union BFrag { v16bf v; unsigned int u[8]; uint4 q[2]; };

// round-to-nearest-even f32->bf16, packed pair into one dword
__device__ __forceinline__ unsigned int pk2bf(float a, float b) {
  union { float f; unsigned int u; } x, y;
  x.f = a; y.f = b;
  unsigned int ua = x.u + (0x7fffu + ((x.u >> 16) & 1u));
  unsigned int ub = y.u + (0x7fffu + ((y.u >> 16) & 1u));
  return (ua >> 16) | (ub & 0xffff0000u);
}

__device__ __forceinline__ float im2col_fetch(const float* __restrict__ B, int gk, int gn,
                                              int K, int N, int H, int W) {
  bool ok = (gk < K) && (gn < N);
  int ci = gk / 9, r = gk - ci * 9;
  int dy = r / 3 - 1, dx = r - (r / 3) * 3 - 1;
  int yy = gn / W, xx = gn - yy * W;
  int y = yy + dy, x = xx + dx;
  ok = ok && (y >= 0) && (y < H) && (x >= 0) && (x < W);
  long idx = ok ? ((long)ci * H * W + (long)y * W + x) : 0;
  float v = B[idx];
  return ok ? v : 0.f;
}

// ---------------------------------------------------------------------------
// Generic GEMM: C[M,N] = A[M,K] * B(K,N), bf16 operands, f32 accumulate.
//   mode 0: B element (k,n) = B[k*ldb + n]         (activations [C,N])
//   mode 1: B element (k,n) = B[n*ldb + k]         (transposed: Q*K^T)
//   mode 2: implicit im2col 3x3: B is [Cin,H,W], K = Cin*9, zero-padded
// Block = 128 threads (4 waves). Block tile 64x64, wave tile 32x32
// (2x2 of 16x16 -> 4 v_wmma per K-step, each fragment reused twice).
// ---------------------------------------------------------------------------
#define AST 40   // LDS row stride in ushorts (80B) -> bank-spread b128 reads

__global__ __launch_bounds__(128)
void gemm_wmma(const float* __restrict__ A, const float* __restrict__ B,
               float* __restrict__ Cc, int M, int K, int N,
               int lda, int ldb, int ldc, int mode, int H, int W) {
  __shared__ __align__(16) unsigned short As[64 * AST];
  __shared__ __align__(16) unsigned short Bs[64 * AST];

  const int t    = threadIdx.x;
  const int lane = t & 31;
  const int wave = t >> 5;
  const int wm   = (wave >> 1) * 32;   // wave M offset in block tile
  const int wn   = (wave & 1) * 32;    // wave N offset in block tile
  const int bm   = blockIdx.y * 64;
  const int bn   = blockIdx.x * 64;

  // staging roles
  const int fm  = t >> 1;              // row 0..63 (A rows / mode-1 B cols)
  const int fkh = (t & 1) * 16;        // k-half within 32
  const int fkp = t >> 3;              // k-pair 0..15 (mode 0/2)
  const int fns = (t & 7) * 8;         // n-segment base (mode 0/2)

  v8f acc00 = {}, acc01 = {}, acc10 = {}, acc11 = {};

  for (int kb = 0; kb < K; kb += 32) {
    const bool fullK = (kb + 32 <= K);
    __syncthreads();

    // ---------------- stage A tile: As[m][k], bf16-pair packed ----------------
    {
      float v[16];
      const int gm = bm + fm;
      if ((bm + 64 <= M) && fullK) {
        const float* src = A + (long)gm * lda + (kb + fkh);
#pragma unroll
        for (int q = 0; q < 4; ++q) {
          f4u r = *(const f4u*)(src + q * 4);
          v[q * 4 + 0] = r.x; v[q * 4 + 1] = r.y; v[q * 4 + 2] = r.z; v[q * 4 + 3] = r.w;
        }
      } else {
#pragma unroll
        for (int j = 0; j < 16; ++j) {
          int gk = kb + fkh + j;
          bool ok = (gm < M) && (gk < K);
          long idx = ok ? ((long)gm * lda + gk) : 0;
          float x = A[idx];
          v[j] = ok ? x : 0.f;
        }
      }
      uint4 p0, p1;
      p0.x = pk2bf(v[0], v[1]);   p0.y = pk2bf(v[2], v[3]);
      p0.z = pk2bf(v[4], v[5]);   p0.w = pk2bf(v[6], v[7]);
      p1.x = pk2bf(v[8], v[9]);   p1.y = pk2bf(v[10], v[11]);
      p1.z = pk2bf(v[12], v[13]); p1.w = pk2bf(v[14], v[15]);
      *(uint4*)&As[fm * AST + fkh]     = p0;
      *(uint4*)&As[fm * AST + fkh + 8] = p1;
    }

    // ---------------- stage B tile: Bs[n][k] (K-contiguous) ----------------
    if (mode == 1) {
      float v[16];
      const int gn = bn + fm;
      if ((bn + 64 <= N) && fullK) {
        const float* src = B + (long)gn * ldb + (kb + fkh);
#pragma unroll
        for (int q = 0; q < 4; ++q) {
          f4u r = *(const f4u*)(src + q * 4);
          v[q * 4 + 0] = r.x; v[q * 4 + 1] = r.y; v[q * 4 + 2] = r.z; v[q * 4 + 3] = r.w;
        }
      } else {
#pragma unroll
        for (int j = 0; j < 16; ++j) {
          int gk = kb + fkh + j;
          bool ok = (gn < N) && (gk < K);
          long idx = ok ? ((long)gn * ldb + gk) : 0;
          float x = B[idx];
          v[j] = ok ? x : 0.f;
        }
      }
      uint4 p0, p1;
      p0.x = pk2bf(v[0], v[1]);   p0.y = pk2bf(v[2], v[3]);
      p0.z = pk2bf(v[4], v[5]);   p0.w = pk2bf(v[6], v[7]);
      p1.x = pk2bf(v[8], v[9]);   p1.y = pk2bf(v[10], v[11]);
      p1.z = pk2bf(v[12], v[13]); p1.w = pk2bf(v[14], v[15]);
      *(uint4*)&Bs[fm * AST + fkh]     = p0;
      *(uint4*)&Bs[fm * AST + fkh + 8] = p1;
    } else {
      // two k-rows (k0, k0+1) x 8 n values, packed per-n into one dword
      float v0[8], v1[8];
      const int k0  = fkp * 2;
      const int gk0 = kb + k0, gk1 = gk0 + 1;
      const int gn  = bn + fns;
      if (mode == 0) {
        if ((bn + 64 <= N) && fullK) {
          const float* s0 = B + (long)gk0 * ldb + gn;
          const float* s1 = B + (long)gk1 * ldb + gn;
          f4u r0 = *(const f4u*)(s0), r1 = *(const f4u*)(s0 + 4);
          f4u r2 = *(const f4u*)(s1), r3 = *(const f4u*)(s1 + 4);
          v0[0] = r0.x; v0[1] = r0.y; v0[2] = r0.z; v0[3] = r0.w;
          v0[4] = r1.x; v0[5] = r1.y; v0[6] = r1.z; v0[7] = r1.w;
          v1[0] = r2.x; v1[1] = r2.y; v1[2] = r2.z; v1[3] = r2.w;
          v1[4] = r3.x; v1[5] = r3.y; v1[6] = r3.z; v1[7] = r3.w;
        } else {
#pragma unroll
          for (int j = 0; j < 8; ++j) {
            int gnj = gn + j;
            bool ok0 = (gk0 < K) && (gnj < N);
            bool ok1 = (gk1 < K) && (gnj < N);
            long i0 = ok0 ? ((long)gk0 * ldb + gnj) : 0;
            long i1 = ok1 ? ((long)gk1 * ldb + gnj) : 0;
            float x0 = B[i0], x1 = B[i1];
            v0[j] = ok0 ? x0 : 0.f;
            v1[j] = ok1 ? x1 : 0.f;
          }
        }
      } else {  // mode 2: implicit im2col 3x3
#pragma unroll
        for (int j = 0; j < 8; ++j) {
          v0[j] = im2col_fetch(B, gk0, gn + j, K, N, H, W);
          v1[j] = im2col_fetch(B, gk1, gn + j, K, N, H, W);
        }
      }
#pragma unroll
      for (int j = 0; j < 8; ++j)
        *(unsigned int*)&Bs[(fns + j) * AST + k0] = pk2bf(v0[j], v1[j]);
    }
    __syncthreads();

    // ---------------- fragments + 4x WMMA ----------------
    BFrag a0, a1, b0, b1;
    const int r15   = lane & 15;
    const int khalf = lane >> 4;
    {
      const unsigned short* ab = &As[(wm + r15) * AST + khalf * 8];
      a0.q[0] = *(const uint4*)(ab);
      a0.q[1] = *(const uint4*)(ab + 16);
      const unsigned short* ab1 = ab + 16 * AST;
      a1.q[0] = *(const uint4*)(ab1);
      a1.q[1] = *(const uint4*)(ab1 + 16);
      const unsigned short* bb = &Bs[(wn + r15) * AST + khalf * 8];
      b0.q[0] = *(const uint4*)(bb);
      b0.q[1] = *(const uint4*)(bb + 16);
      const unsigned short* bb1 = bb + 16 * AST;
      b1.q[0] = *(const uint4*)(bb1);
      b1.q[1] = *(const uint4*)(bb1 + 16);
    }
    acc00 = __builtin_amdgcn_wmma_f32_16x16x32_bf16(false, a0.v, false, b0.v, (short)0, acc00, false, false);
    acc01 = __builtin_amdgcn_wmma_f32_16x16x32_bf16(false, a0.v, false, b1.v, (short)0, acc01, false, false);
    acc10 = __builtin_amdgcn_wmma_f32_16x16x32_bf16(false, a1.v, false, b0.v, (short)0, acc10, false, false);
    acc11 = __builtin_amdgcn_wmma_f32_16x16x32_bf16(false, a1.v, false, b1.v, (short)0, acc11, false, false);
  }

  // ---------------- epilogue: f32 16x16 C/D layout ----------------
  const int r15 = lane & 15;
  const int mh  = (lane >> 4) * 8;
#pragma unroll
  for (int i = 0; i < 2; ++i) {
#pragma unroll
    for (int j = 0; j < 2; ++j) {
      v8f acc = (i == 0) ? (j == 0 ? acc00 : acc01) : (j == 0 ? acc10 : acc11);
      int n = bn + wn + j * 16 + r15;
      int mb = bm + wm + i * 16 + mh;
      if (n < N) {
#pragma unroll
        for (int r = 0; r < 8; ++r) {
          int m = mb + r;
          if (m < M) Cc[(long)m * ldc + n] = acc[r];
        }
      }
    }
  }
}

// ---------------------------------------------------------------------------
// Support kernels
// ---------------------------------------------------------------------------
__global__ void k_layernorm_c(const float* __restrict__ X, const float* __restrict__ w,
                              const float* __restrict__ b, float* __restrict__ Y,
                              int C, int N) {
  int n = blockIdx.x * blockDim.x + threadIdx.x;
  if (n >= N) return;
  float mu = 0.f;
  for (int c = 0; c < C; ++c) mu += X[(long)c * N + n];
  mu /= (float)C;
  float var = 0.f;
  for (int c = 0; c < C; ++c) { float d = X[(long)c * N + n] - mu; var += d * d; }
  var /= (float)C;
  float inv = rsqrtf(var + 1e-5f);
  for (int c = 0; c < C; ++c)
    Y[(long)c * N + n] = (X[(long)c * N + n] - mu) * inv * w[c] + b[c];
}

__global__ void k_dwconv3x3(const float* __restrict__ in, const float* __restrict__ w,
                            float* __restrict__ out, int C, int H, int W) {
  long i = (long)blockIdx.x * blockDim.x + threadIdx.x;
  long total = (long)C * H * W;
  if (i >= total) return;
  int x = (int)(i % W);
  int y = (int)((i / W) % H);
  int c = (int)(i / ((long)H * W));
  const float* wc = w + (long)c * 9;
  float s = 0.f;
#pragma unroll
  for (int dy = -1; dy <= 1; ++dy)
#pragma unroll
    for (int dx = -1; dx <= 1; ++dx) {
      int yy = y + dy, xx = x + dx;
      if (yy >= 0 && yy < H && xx >= 0 && xx < W)
        s += wc[(dy + 1) * 3 + (dx + 1)] * in[(long)c * H * W + (long)yy * W + xx];
    }
  out[i] = s;
}

__global__ void k_l2norm_rows(float* __restrict__ X, int N) {
  __shared__ float red[256];
  float* row = X + (long)blockIdx.x * N;
  float s = 0.f;
  for (int i = threadIdx.x; i < N; i += blockDim.x) { float v = row[i]; s += v * v; }
  red[threadIdx.x] = s;
  __syncthreads();
  for (int st = 128; st > 0; st >>= 1) {
    if ((int)threadIdx.x < st) red[threadIdx.x] += red[threadIdx.x + st];
    __syncthreads();
  }
  float inv = 1.f / fmaxf(sqrtf(red[0]), 1e-12f);
  for (int i = threadIdx.x; i < N; i += blockDim.x) row[i] *= inv;
}

__global__ void k_softmax_rows(float* __restrict__ S, const float* __restrict__ temp,
                               int dh) {
  int h = blockIdx.x;
  int r = threadIdx.x;
  if (r >= dh) return;
  float t = temp[h];
  float* row = S + (long)h * dh * dh + (long)r * dh;
  float mx = -3.4e38f;
  for (int j = 0; j < dh; ++j) { float v = row[j] * t; mx = fmaxf(mx, v); }
  float sum = 0.f;
  for (int j = 0; j < dh; ++j) { float e = __expf(row[j] * t - mx); row[j] = e; sum += e; }
  float inv = 1.f / sum;
  for (int j = 0; j < dh; ++j) row[j] *= inv;
}

__global__ void k_add2_scaled(float* __restrict__ X, const float* __restrict__ A,
                              const float* __restrict__ B, long n, float s) {
  long i = (long)blockIdx.x * blockDim.x + threadIdx.x;
  if (i < n) X[i] = X[i] + (A[i] + B[i]) * s;
}

__global__ void k_axpy(float* __restrict__ X, const float* __restrict__ T, long n, float s) {
  long i = (long)blockIdx.x * blockDim.x + threadIdx.x;
  if (i < n) X[i] = X[i] + T[i] * s;
}

__global__ void k_hswish_gate(const float* __restrict__ Y, float* __restrict__ G,
                              int hc, int N) {
  long i = (long)blockIdx.x * blockDim.x + threadIdx.x;
  long total = (long)hc * N;
  if (i >= total) return;
  float x1 = Y[i];
  float x2 = Y[total + i];
  float hs = x1 * fminf(fmaxf(x1 + 3.f, 0.f), 6.f) * (1.f / 6.f);
  G[i] = hs * x2;
}

__global__ void k_sigmoid_gate(const float* __restrict__ CT, const float* __restrict__ AL,
                               float* __restrict__ G, int C, int N) {
  long i = (long)blockIdx.x * blockDim.x + threadIdx.x;
  long total = (long)C * N;
  if (i >= total) return;
  float e = CT[i];
  float d = CT[total + i];
  float a = AL[i];
  float sg = 1.f / (1.f + __expf(-a));
  G[i] = d + e * sg;
}

__global__ void k_unshuffle2(const float* __restrict__ in, float* __restrict__ out,
                             int C, int H, int W) {  // in [C,H,W] -> out [4C,H/2,W/2]
  long i = (long)blockIdx.x * blockDim.x + threadIdx.x;
  int Ho = H >> 1, Wo = W >> 1;
  long total = (long)C * H * W;
  if (i >= total) return;
  int xo = (int)(i % Wo);
  int yo = (int)((i / Wo) % Ho);
  int co = (int)(i / ((long)Ho * Wo));
  int c  = co >> 2, rr = co & 3, r1 = rr >> 1, r2 = rr & 1;
  out[i] = in[(long)c * H * W + (long)(2 * yo + r1) * W + (2 * xo + r2)];
}

__global__ void k_shuffle2(const float* __restrict__ in, float* __restrict__ out,
                           int C, int H, int W) {  // in [C,H,W] -> out [C/4,2H,2W]
  long i = (long)blockIdx.x * blockDim.x + threadIdx.x;
  int Ho = H << 1, Wo = W << 1;
  long total = (long)C * H * W;
  if (i >= total) return;
  int x = (int)(i % Wo);
  int y = (int)((i / Wo) % Ho);
  int co = (int)(i / ((long)Ho * Wo));
  int r1 = y & 1, r2 = x & 1;
  out[i] = in[(long)(co * 4 + r1 * 2 + r2) * H * W + (long)(y >> 1) * W + (x >> 1)];
}

__global__ void k_final(const float* __restrict__ F, const float* __restrict__ ob,
                        const float* __restrict__ xin, float* __restrict__ out, int N) {
  long i = (long)blockIdx.x * blockDim.x + threadIdx.x;
  if (i >= (long)3 * N) return;
  out[i] = F[i] + ob[i / N] + xin[i];
}

// ---------------------------------------------------------------------------
// Host-side orchestration
// ---------------------------------------------------------------------------
struct Mdta { const float *dw, *proj, *qkv, *temp; };
struct Gdfn { const float *dw, *pin, *pout; };
struct Blk  { Gdfn ffn; const float *n1b, *n1w, *n2b, *n2w; Mdta s[2]; };
struct Csaf { const float *attn, *dp, *ep, *out; };

struct Cursor {
  void* const* d; int i;
  const float* next() { return (const float*)d[i++]; }
};

static Blk parseBlk(Cursor& c) {   // jax-sorted keys: ffn{dw,pin,pout}, n1b,n1w,n2b,n2w, streams[2]{dw,proj,qkv,temp}
  Blk b;
  b.ffn.dw = c.next(); b.ffn.pin = c.next(); b.ffn.pout = c.next();
  b.n1b = c.next(); b.n1w = c.next(); b.n2b = c.next(); b.n2w = c.next();
  for (int s = 0; s < 2; ++s) {
    b.s[s].dw = c.next(); b.s[s].proj = c.next(); b.s[s].qkv = c.next(); b.s[s].temp = c.next();
  }
  return b;
}
static Csaf parseCsaf(Cursor& c) { // sorted: attn, dp, ep, out
  Csaf p; p.attn = c.next(); p.dp = c.next(); p.ep = c.next(); p.out = c.next(); return p;
}

static inline int cdiv(long a, int b) { return (int)((a + b - 1) / b); }

struct Ctx {
  hipStream_t st;
  float *XN, *Yb, *A1, *A2, *B1, *B2, *Sb, *Ub, *CT, *AL;
};

static void gemm(const Ctx& c, const float* A, const float* B, float* C,
                 int M, int K, int N, int lda, int ldb, int ldc,
                 int mode, int H, int W) {
  dim3 g((N + 63) / 64, (M + 63) / 64, 1);
  gemm_wmma<<<g, 128, 0, c.st>>>(A, B, C, M, K, N, lda, ldb, ldc, mode, H, W);
}

static void run_block(const Ctx& c, float* X, const Blk& bp, int C, int heads, int H, int W) {
  const int N  = H * W;
  const int dh = C / heads;                  // 48, except 96 for C=96 with 1 head
  const int hc = (int)((double)C * 2.66);
  const long CN = (long)C * N;

  // LN1
  k_layernorm_c<<<cdiv(N, 256), 256, 0, c.st>>>(X, bp.n1w, bp.n1b, c.XN, C, N);

  // two MDTA streams
  for (int s = 0; s < 2; ++s) {
    // qkv = 1x1 conv + depthwise 3x3
    gemm(c, bp.s[s].qkv, c.XN, c.B1, 3 * C, C, N, C, N, N, 0, H, W);
    k_dwconv3x3<<<cdiv((long)3 * C * N, 256), 256, 0, c.st>>>(c.B1, bp.s[s].dw, c.B2, 3 * C, H, W);
    // l2-normalize q,k rows (first 2C rows of B2)
    k_l2norm_rows<<<2 * C, 256, 0, c.st>>>(c.B2, N);
    // per-head S = Qh * Kh^T   (M=dh, K=N, N=dh)
    for (int h = 0; h < heads; ++h) {
      const float* Q = c.B2 + (long)(h * dh) * N;
      const float* Kp = c.B2 + (long)(C + h * dh) * N;
      gemm(c, Q, Kp, c.Sb + (long)h * dh * dh, dh, N, dh, N, N, dh, 1, H, W);
    }
    k_softmax_rows<<<heads, 64, 0, c.st>>>(c.Sb, bp.s[s].temp, dh);
    // per-head out = S * Vh
    for (int h = 0; h < heads; ++h) {
      const float* V = c.B2 + (long)(2 * C + h * dh) * N;
      gemm(c, c.Sb + (long)h * dh * dh, V, c.Yb + (long)(h * dh) * N,
           dh, dh, N, dh, N, N, 0, H, W);
    }
    // proj 1x1
    gemm(c, bp.s[s].proj, c.Yb, (s == 0 ? c.A1 : c.A2), C, C, N, C, N, N, 0, H, W);
  }
  // x += (A1 + A2) * 0.5 * 0.1
  k_add2_scaled<<<cdiv(CN, 256), 256, 0, c.st>>>(X, c.A1, c.A2, CN, 0.05f);

  // GDFN
  k_layernorm_c<<<cdiv(N, 256), 256, 0, c.st>>>(X, bp.n2w, bp.n2b, c.XN, C, N);
  gemm(c, bp.ffn.pin, c.XN, c.B1, 2 * hc, C, N, C, N, N, 0, H, W);
  k_dwconv3x3<<<cdiv((long)2 * hc * N, 256), 256, 0, c.st>>>(c.B1, bp.ffn.dw, c.B2, 2 * hc, H, W);
  k_hswish_gate<<<cdiv((long)hc * N, 256), 256, 0, c.st>>>(c.B2, c.B1, hc, N);
  gemm(c, bp.ffn.pout, c.B1, c.A1, C, hc, N, hc, N, N, 0, H, W);
  k_axpy<<<cdiv(CN, 256), 256, 0, c.st>>>(X, c.A1, CN, 0.1f);
}

static void run_csaf(const Ctx& c, const float* enc, const float* dec, const Csaf& p,
                     int C, int H, int W, int om, float* out) {
  const int N = H * W;
  const long CN = (long)C * N;
  gemm(c, p.ep, enc, c.CT,            C, C, N, C, N, N, 0, H, W);   // e -> rows [0,C)
  gemm(c, p.dp, dec, c.CT + CN,       C, C, N, C, N, N, 0, H, W);   // d -> rows [C,2C)
  gemm(c, p.attn, c.CT, c.AL,         C, 2 * C, N, 2 * C, N, N, 0, H, W);
  k_sigmoid_gate<<<cdiv(CN, 256), 256, 0, c.st>>>(c.CT, c.AL, c.Ub, C, N);
  gemm(c, p.out, c.Ub, out,           C * om, C, N, C, N, N, 0, H, W);
}

extern "C" void kernel_launch(void* const* d_in, const int* in_sizes, int n_in,
                              void* d_out, int out_size, void* d_ws, size_t ws_size,
                              hipStream_t stream) {
  if (n_in < 682) return;   // 681 param leaves + x

  // x detected by element count (1*3*128*128); params in jax tree (sorted-key) order
  const float* x_in = nullptr;
  Cursor cur{d_in, 0};
  bool x_first = (in_sizes[0] == 3 * 128 * 128);
  if (x_first) { x_in = (const float*)d_in[0]; cur.i = 1; }

  Csaf csa0 = parseCsaf(cur);        // c=192
  Csaf csa1 = parseCsaf(cur);        // c=96
  Csaf csa_bn = parseCsaf(cur);      // c=48, om=2
  Blk dec0[6]; for (int i = 0; i < 6; ++i) dec0[i] = parseBlk(cur);
  Blk dec1[6]; for (int i = 0; i < 6; ++i) dec1[i] = parseBlk(cur);
  Blk dec2[4]; for (int i = 0; i < 4; ++i) dec2[i] = parseBlk(cur);
  const float* downs[3]; for (int i = 0; i < 3; ++i) downs[i] = cur.next();
  const float* embed = cur.next();
  Blk enc0[4]; for (int i = 0; i < 4; ++i) enc0[i] = parseBlk(cur);
  Blk enc1[6]; for (int i = 0; i < 6; ++i) enc1[i] = parseBlk(cur);
  Blk enc2[6]; for (int i = 0; i < 6; ++i) enc2[i] = parseBlk(cur);
  Blk enc3[8]; for (int i = 0; i < 8; ++i) enc3[i] = parseBlk(cur);
  const float* out_b = cur.next();
  const float* out_w = cur.next();
  Blk ref[4]; for (int i = 0; i < 4; ++i) ref[i] = parseBlk(cur);
  const float* ups[3]; for (int i = 0; i < 3; ++i) ups[i] = cur.next();
  if (!x_first) x_in = (const float*)d_in[cur.i];

  // ---- workspace bump allocator (fp32 buffers) ----
  char* wsb = (char*)d_ws;
  size_t off = 0;
  auto alloc = [&](size_t floats) -> float* {
    float* p = (float*)(wsb + off);
    off += ((floats * 4) + 255) & ~(size_t)255;
    return p;
  };
  float* E1 = alloc((size_t)48 * 16384);       // level-0 skip
  float* E2 = alloc((size_t)96 * 4096);        // level-1 skip
  float* E3 = alloc((size_t)192 * 1024);       // level-2 skip
  float* Xb = alloc((size_t)96 * 16384);       // current activations (max C*N)
  Ctx c;
  c.st = stream;
  c.XN = alloc((size_t)96 * 16384);
  c.Yb = alloc((size_t)96 * 16384);
  c.A1 = alloc((size_t)96 * 16384);
  c.A2 = alloc((size_t)96 * 16384);
  c.B1 = alloc((size_t)510 * 16384);           // max 2*hc*N
  c.B2 = alloc((size_t)510 * 16384);
  c.Ub = alloc((size_t)192 * 4096);            // conv-before-(un)shuffle / csaf gate
  c.CT = alloc((size_t)96 * 16384);            // csaf concat [2C,N]
  c.AL = alloc((size_t)48 * 16384);            // csaf attention logits
  c.Sb = alloc((size_t)8 * 48 * 48);           // per-head attention matrices
  (void)ws_size; (void)out_size;

  // ---- encoder ----
  gemm(c, embed, x_in, E1, 48, 27, 16384, 27, 0, 16384, 2, 128, 128);
  for (int i = 0; i < 4; ++i) run_block(c, E1, enc0[i], 48, 1, 128, 128);

  gemm(c, downs[0], E1, c.Ub, 24, 48 * 9, 16384, 48 * 9, 0, 16384, 2, 128, 128);
  k_unshuffle2<<<cdiv((long)24 * 16384, 256), 256, 0, stream>>>(c.Ub, E2, 24, 128, 128);
  for (int i = 0; i < 6; ++i) run_block(c, E2, enc1[i], 96, 2, 64, 64);

  gemm(c, downs[1], E2, c.Ub, 48, 96 * 9, 4096, 96 * 9, 0, 4096, 2, 64, 64);
  k_unshuffle2<<<cdiv((long)48 * 4096, 256), 256, 0, stream>>>(c.Ub, E3, 48, 64, 64);
  for (int i = 0; i < 6; ++i) run_block(c, E3, enc2[i], 192, 4, 32, 32);

  gemm(c, downs[2], E3, c.Ub, 96, 192 * 9, 1024, 192 * 9, 0, 1024, 2, 32, 32);
  k_unshuffle2<<<cdiv((long)96 * 1024, 256), 256, 0, stream>>>(c.Ub, Xb, 96, 32, 32);
  for (int i = 0; i < 8; ++i) run_block(c, Xb, enc3[i], 384, 8, 16, 16);   // e4 in Xb

  // ---- decoder ----
  gemm(c, ups[0], Xb, c.Ub, 768, 384 * 9, 256, 384 * 9, 0, 256, 2, 16, 16);
  k_shuffle2<<<cdiv((long)768 * 256, 256), 256, 0, stream>>>(c.Ub, c.Yb, 768, 16, 16);
  run_csaf(c, E3, c.Yb, csa0, 192, 32, 32, 1, Xb);
  for (int i = 0; i < 6; ++i) run_block(c, Xb, dec0[i], 192, 4, 32, 32);

  gemm(c, ups[1], Xb, c.Ub, 384, 192 * 9, 1024, 192 * 9, 0, 1024, 2, 32, 32);
  k_shuffle2<<<cdiv((long)384 * 1024, 256), 256, 0, stream>>>(c.Ub, c.Yb, 384, 32, 32);
  run_csaf(c, E2, c.Yb, csa1, 96, 64, 64, 1, Xb);
  for (int i = 0; i < 6; ++i) run_block(c, Xb, dec1[i], 96, 2, 64, 64);

  gemm(c, ups[2], Xb, c.Ub, 192, 96 * 9, 4096, 96 * 9, 0, 4096, 2, 64, 64);
  k_shuffle2<<<cdiv((long)192 * 4096, 256), 256, 0, stream>>>(c.Ub, c.Yb, 192, 64, 64);
  run_csaf(c, E1, c.Yb, csa_bn, 48, 128, 128, 2, Xb);                      // -> [96,16384]
  for (int i = 0; i < 4; ++i) run_block(c, Xb, dec2[i], 96, 1, 128, 128);

  // ---- refinement + output ----
  for (int i = 0; i < 4; ++i) run_block(c, Xb, ref[i], 96, 1, 128, 128);
  gemm(c, out_w, Xb, c.XN, 3, 96 * 9, 16384, 96 * 9, 0, 16384, 2, 128, 128);
  k_final<<<cdiv((long)3 * 16384, 256), 256, 0, stream>>>(c.XN, out_b, x_in, (float*)d_out, 16384);
}
